// DifferentiableDenseHGTConv_39865886442168
// MI455X (gfx1250) — matrix operand order repaired and checked by reading, not compiled, not run on hardware
//
#include <hip/hip_runtime.h>
#include <hip/hip_bf16.h>
#include <math.h>

#define TT   3
#define RR   4
#define HH   8
#define DKK  32
#define DOUT 256
#define DIN  256
#define NB   4
#define NN   1024

#ifndef __has_builtin
#define __has_builtin(x) 0
#endif
#define USE_TDM __has_builtin(__builtin_amdgcn_tensor_load_to_lds)

typedef __attribute__((ext_vector_type(16))) __bf16 v16bf;
typedef __attribute__((ext_vector_type(8)))  float  v8f;
typedef unsigned int __attribute__((ext_vector_type(4))) v4u;
typedef int  __attribute__((ext_vector_type(8))) v8i_;
typedef int  __attribute__((ext_vector_type(4))) v4i_;

__device__ __forceinline__ int lane_id() { return (int)(threadIdx.x & 31u); }

// WMMA 16x16x32 bf16 operand layouts used by this code (contiguous k, A and B alike):
// A (16Mx32K): m=lane&15, hi=lane>>4; elem e -> k=hi*16+e
// B (32Kx16N): n=lane&15; k = (lane>>4)*16 + e
// C (16Mx16N f32): per v: m = v + (lane>>4)*8 ; n = lane&15
__device__ __forceinline__ int a_k(int e, int hi) {
  return hi * 16 + e;
}

__device__ __forceinline__ v16bf gatherA_f32(const float* t, int ld) {
  int l = lane_id(), m = l & 15, hi = l >> 4;
  v16bf a;
#pragma unroll
  for (int e = 0; e < 16; ++e) a[e] = (__bf16)t[(size_t)m * ld + a_k(e, hi)];
  return a;
}
__device__ __forceinline__ v16bf gatherA_bf(const __bf16* t, int ld) {
  int l = lane_id(), m = l & 15, hi = l >> 4;
  v16bf a;
#pragma unroll
  for (int e = 0; e < 16; ++e) a[e] = t[(size_t)m * ld + a_k(e, hi)];
  return a;
}
// B gather from row-major [K][N] f32 tile
__device__ __forceinline__ v16bf gatherB_f32(const float* t, int ld) {
  int l = lane_id(), n = l & 15, k0 = (l >> 4) * 16;
  v16bf b;
#pragma unroll
  for (int e = 0; e < 16; ++e) b[e] = (__bf16)t[(size_t)(k0 + e) * ld + n];
  return b;
}
// B gather from [N][K] f32 tile (transposed source)
__device__ __forceinline__ v16bf gatherBT_f32(const float* t, int ld) {
  int l = lane_id(), n = l & 15, k0 = (l >> 4) * 16;
  v16bf b;
#pragma unroll
  for (int e = 0; e < 16; ++e) b[e] = (__bf16)t[(size_t)n * ld + k0 + e];
  return b;
}

// fragments stored lane-major: 32 lanes x 16 bf16 contiguous (512 elems, 1KB)
__device__ __forceinline__ v16bf ldfrag(const __bf16* p) {
  return *(const v16bf*)(p + (size_t)lane_id() * 16);
}
__device__ __forceinline__ void stfrag(__bf16* p, v16bf v) {
  *(v16bf*)(p + (size_t)lane_id() * 16) = v;
}

__device__ __forceinline__ v8f wmma_bf16(v16bf a, v16bf b, v8f c) {
  return __builtin_amdgcn_wmma_f32_16x16x32_bf16(false, a, false, b, (short)0, c, false, false);
}

__device__ __forceinline__ void lds_fence() {
  asm volatile("s_wait_dscnt 0" ::: "memory");
}

__device__ __forceinline__ float red16_add(float x) {
#pragma unroll
  for (int o = 1; o < 16; o <<= 1) x += __shfl_xor(x, o, 32);
  return x;
}
__device__ __forceinline__ float red16_max(float x) {
#pragma unroll
  for (int o = 1; o < 16; o <<= 1) x = fmaxf(x, __shfl_xor(x, o, 32));
  return x;
}

#if USE_TDM
// Tensor Data Mover: async DMA of a 2D tile (tileX contiguous f32 per row,
// tileY rows, row stride in elements) from global memory into LDS.
// D# bit layout per CDNA5 ISA ch.8 (group0 128b, group1 256b, groups 2/3 zero -> 2D).
// This toolchain exposes the 6-arg builtin: (v4u, v8i, v4i, v4i, v8i, i32 cpol).
__device__ __forceinline__ void tdm_load_2d(float* ldsDst, const float* gsrc,
                                            unsigned tileX, unsigned tileY,
                                            long long strideElems) {
  unsigned long long ga = (unsigned long long)(const void*)gsrc;
  unsigned ldso = (unsigned)(unsigned long long)(void*)ldsDst;  // LDS byte offset
  unsigned long long s0 = (unsigned long long)strideElems;
  const unsigned d0 = 1u << 20, d1 = 1u << 20;  // generous tensor dims (no OOB)
  v4u g0 = { 1u,                                   // count=1, user mode
             ldso,                                 // lds_addr
             (unsigned)ga,                         // global_addr[31:0]
             (unsigned)((ga >> 32) & 0x01ffffffu) | (2u << 30) };  // addr[56:32] | type=2
  v8i_ g1 = { (int)(2u << 16),                                    // data_size=4B
              (int)((d0 & 0xffffu) << 16),                        // tensor_dim0 lo16
              (int)((d0 >> 16) | ((d1 & 0xffffu) << 16)),         // dim0 hi | dim1 lo
              (int)((d1 >> 16) | (tileX << 16)),                  // dim1 hi | tile_dim0
              (int)(tileY & 0xffffu),                             // tile_dim1, tile_dim2=0
              (int)(unsigned)(s0 & 0xffffffffull),                // dim0_stride[31:0]
              (int)(unsigned)((s0 >> 32) & 0xffffull),            // dim0_stride[47:32]
              0 };
  v4i_ g2 = { 0, 0, 0, 0 };
  v4i_ g3 = { 0, 0, 0, 0 };
  v8i_ g4 = { 0, 0, 0, 0, 0, 0, 0, 0 };
  __builtin_amdgcn_tensor_load_to_lds(g0, g1, g2, g3, g4, 0);
}
#endif

// ---------------------------------------------------------------------------
// Kernel 1: per-type Q/K/V projections.  grid(64,B), 512 thr (16 waves, wave=o-tile)
// One 16x16 output tile per call; explicit weight pointers keep global
// provenance without runtime pointer selects; t-loop kept rolled to bound
// register pressure (no spills).
// ---------------------------------------------------------------------------
__device__ __forceinline__ void proj_one(const float* xs, const float* __restrict__ W,
                                         const float* __restrict__ bvec,
                                         float* __restrict__ dstM,
                                         int t, int bb, int n0, int o0) {
  int l = lane_id(), n = l & 15, hi = l >> 4;
  v8f c = {};
#pragma unroll
  for (int kc = 0; kc < 8; ++kc) {
    v16bf a = gatherA_f32(xs + kc * 32, DIN);
    v16bf b = gatherB_f32(W + (size_t)t * DIN * DOUT + (size_t)kc * 32 * DOUT + o0, DOUT);
    c = wmma_bf16(a, b, c);
  }
  float bias = bvec[t * DOUT + o0 + n];
  float* dst = dstM + ((size_t)(bb * TT + t) * NN + n0) * DOUT + o0;
#pragma unroll
  for (int v = 0; v < 8; ++v)
    dst[(size_t)(v + hi * 8) * DOUT + n] = c[v] + bias;
}

__global__ __launch_bounds__(512)
void k_qkv(const float* __restrict__ x,
           const float* __restrict__ Wq, const float* __restrict__ bq,
           const float* __restrict__ Wk, const float* __restrict__ bk,
           const float* __restrict__ Wv, const float* __restrict__ bv,
           float* __restrict__ qt, float* __restrict__ kt, float* __restrict__ vt) {
  __shared__ float xs[16 * DIN];
  int bb = blockIdx.y, n0 = blockIdx.x * 16;
  int tid = threadIdx.x;
  for (int i = tid; i < 16 * DIN; i += 512)
    xs[i] = x[((size_t)bb * NN + n0 + (i >> 8)) * DIN + (i & 255)];
  __syncthreads();
  int o0 = (tid >> 5) * 16;
#pragma unroll 1
  for (int t = 0; t < TT; ++t) {
    proj_one(xs, Wq, bq, qt, t, bb, n0, o0);
    proj_one(xs, Wk, bk, kt, t, bb, n0, o0);
    proj_one(xs, Wv, bv, vt, t, bb, n0, o0);
  }
}

// ---------------------------------------------------------------------------
// Kernel 2: Ksr/Vsr relation transforms + Qbar/Kbar soft-type mixes.
// Writes attention operands to global in WMMA fragment order.
// K path and V path processed sequentially to halve live register state.
// grid(32,B) (32-row tiles), 256 thr (8 waves, wave=head)
// ---------------------------------------------------------------------------
__global__ __launch_bounds__(256)
void k_relmix(const float* __restrict__ qt, const float* __restrict__ kt,
              const float* __restrict__ vt, const float* __restrict__ types,
              const float* __restrict__ rel_att, const float* __restrict__ rel_msg,
              __bf16* __restrict__ qbarF, __bf16* __restrict__ kbarF,
              __bf16* __restrict__ ksrF, __bf16* __restrict__ vsrF) {
  __shared__ float wt[32][TT];
  __shared__ float sc[HH][32 * 32];
  int bb = blockIdx.y, nt = blockIdx.x, n0 = nt * 32;
  int tid = threadIdx.x;
  if (tid < 32 * TT)
    wt[tid / TT][tid % TT] = types[((size_t)bb * NN + n0 + tid / TT) * TT + tid % TT];
  __syncthreads();
  int h = tid >> 5;
  int l = lane_id(), n = l & 15, hi = l >> 4;
  float* mysc = sc[h];

#pragma unroll 1
  for (int r = 0; r < RR; ++r) {
    v8f kb[2][2] = {};
#pragma unroll 1
    for (int t = 0; t < TT; ++t) {
      // ---- K path: Ksr = Kt @ rel_att[r,h], accumulate Kbar ----
      {
#pragma unroll 1
        for (int u = 0; u < 2; ++u) {
          v16bf KtA = gatherA_f32(
              kt + ((size_t)(bb * TT + t) * NN + n0 + 16 * u) * DOUT + h * DKK, DOUT);
#pragma unroll
          for (int eh = 0; eh < 2; ++eh) {
            v16bf RaB = gatherB_f32(rel_att + ((size_t)(r * HH + h) * 32) * 32 + eh * 16, 32);
            v8f z = {};
            v8f ks = wmma_bf16(KtA, RaB, z);
#pragma unroll
            for (int v = 0; v < 8; ++v) {
              int row = 16 * u + v + hi * 8;
              kb[u][eh][v] += wt[row][t] * ks[v];
              mysc[row * 32 + eh * 16 + n] = ks[v];
            }
          }
        }
        lds_fence();
        // Ksr as logits-B fragments: [K=dk(32) x N=j(16)] per 16-row j-tile
#pragma unroll
        for (int jt = 0; jt < 2; ++jt) {
          v16bf f = gatherBT_f32(mysc + jt * 16 * 32, 32);
          stfrag(ksrF + (((size_t)((bb * TT + t) * RR + r) * HH + h) * 64 +
                         (size_t)nt * 2 + jt) * 512, f);
        }
        lds_fence();
      }
      // ---- V path: Vsr = Vt @ rel_msg[r,h] ----
      {
#pragma unroll 1
        for (int u = 0; u < 2; ++u) {
          v16bf VtA = gatherA_f32(
              vt + ((size_t)(bb * TT + t) * NN + n0 + 16 * u) * DOUT + h * DKK, DOUT);
#pragma unroll
          for (int eh = 0; eh < 2; ++eh) {
            v16bf RmB = gatherB_f32(rel_msg + ((size_t)(r * HH + h) * 32) * 32 + eh * 16, 32);
            v8f z = {};
            v8f vs = wmma_bf16(VtA, RmB, z);
#pragma unroll
            for (int v = 0; v < 8; ++v)
              mysc[(16 * u + v + hi * 8) * 32 + eh * 16 + n] = vs[v];
          }
        }
        lds_fence();
        // Vsr as PV-B fragments: [K=j(32) x N=dk(16)] per dk half
#pragma unroll
        for (int half = 0; half < 2; ++half) {
          v16bf f = gatherB_f32(mysc + half * 16, 32);
          stfrag(vsrF + ((((size_t)((bb * TT + t) * RR + r) * HH + h) * 32 + nt) * 2 +
                         half) * 512, f);
        }
        lds_fence();
      }
    }
    // Kbar fragments
#pragma unroll
    for (int u = 0; u < 2; ++u)
#pragma unroll
      for (int eh = 0; eh < 2; ++eh)
#pragma unroll
        for (int v = 0; v < 8; ++v)
          mysc[(16 * u + v + hi * 8) * 32 + eh * 16 + n] = kb[u][eh][v];
    lds_fence();
#pragma unroll
    for (int jt = 0; jt < 2; ++jt) {
      v16bf f = gatherBT_f32(mysc + jt * 16 * 32, 32);
      stfrag(kbarF + (((size_t)(bb * RR + r) * HH + h) * 64 + (size_t)nt * 2 + jt) * 512, f);
    }
    lds_fence();
  }
  // Qbar A-fragments (elementwise soft-type mix of Qt)
  int m = l & 15;
#pragma unroll 1
  for (int u = 0; u < 2; ++u) {
    v16bf a;
#pragma unroll
    for (int e = 0; e < 16; ++e) {
      int kk = a_k(e, hi), row = 16 * u + m;
      float s = 0.f;
#pragma unroll
      for (int t = 0; t < TT; ++t)
        s += wt[row][t] * qt[((size_t)(bb * TT + t) * NN + n0 + row) * DOUT + h * DKK + kk];
      a[e] = (__bf16)s;
    }
    stfrag(qbarF + (((size_t)(bb * HH + h)) * 64 + (size_t)nt * 2 + u) * 512, a);
  }
}

// ---------------------------------------------------------------------------
// Kernel 3: attention (two-pass flash softmax + double-weighted aggregation).
// adj/erel tiles stream through the Tensor Data Mover, double-buffered in LDS.
// grid(64,B) (16 target rows), 256 thr (8 waves, wave=head)
// ---------------------------------------------------------------------------
__global__ __launch_bounds__(256)
void k_attn(const float* __restrict__ adj, const float* __restrict__ erel,
            const float* __restrict__ types, const float* __restrict__ rel_pri,
            const __bf16* __restrict__ qbarF, const __bf16* __restrict__ kbarF,
            const __bf16* __restrict__ ksrF, const __bf16* __restrict__ vsrF,
            float* __restrict__ aggG) {
  __shared__ float adjT[2][16 * 32];        // raw adj tile, double buffered
  __shared__ float erelT[2][16 * 32 * RR];  // raw erel tile [i][j][r], double buffered
  __shared__ float ts[32][TT];
  __shared__ float pw[HH][16 * 32];
  __shared__ float hsum[HH][16];
  __shared__ float rok[16];
  int bb = blockIdx.y, it = blockIdx.x, i0 = it * 16;
  int tid = threadIdx.x, h = tid >> 5;
  int l = lane_id(), n = l & 15, hi = l >> 4;
  const float isd = 0.17677669529663687f;  // 1/sqrt(DK)
  float pri[RR];
#pragma unroll
  for (int r = 0; r < RR; ++r) pri[r] = rel_pri[r * HH + h] * isd;
  v16bf qf = ldfrag(qbarF + (((size_t)(bb * HH + h)) * 64 + it) * 512);

#define ISSUE_TILES(BUF, JS)                                                        \
  do {                                                                              \
    tdm_load_2d(adjT[BUF], adj + (size_t)bb * NN * NN + (size_t)i0 * NN + (JS)*32,  \
                32u, 16u, (long long)NN);                                           \
    tdm_load_2d(erelT[BUF], erel + (((size_t)bb * NN + i0) * NN + (size_t)(JS)*32) * RR, \
                128u, 16u, (long long)NN * RR);                                     \
  } while (0)

#define STAGE_FB(BUF, JS)                                                           \
  do {                                                                              \
    for (int i = tid; i < 512; i += 256)                                            \
      adjT[BUF][i] = adj[(size_t)bb * NN * NN + (size_t)(i0 + (i >> 5)) * NN +      \
                         (JS)*32 + (i & 31)];                                       \
    for (int i = tid; i < 2048; i += 256)                                           \
      erelT[BUF][i] = erel[(((size_t)bb * NN + i0 + (i >> 7)) * NN + (JS)*32) * RR + \
                           (i & 127)];                                              \
  } while (0)

  float mx[8], sm[8], rs[8];
#pragma unroll
  for (int v = 0; v < 8; ++v) { mx[v] = -3.0e38f; sm[v] = 0.f; rs[v] = 0.f; }

  // ---- pass 1: per-(row,head) max / sum-exp, raw logit row-sum ----
#if USE_TDM
  if ((tid >> 5) == 0) ISSUE_TILES(0, 0);
#else
  STAGE_FB(0, 0);
#endif
  for (int js = 0; js < NN / 32; ++js) {
    int cur = js & 1;
    if (js + 1 < NN / 32) {
#if USE_TDM
      if ((tid >> 5) == 0) ISSUE_TILES((js + 1) & 1, js + 1);
#else
      STAGE_FB((js + 1) & 1, js + 1);
#endif
      if (l == 0)  // pull next step's Kbar fragment stream toward the WGP
        __builtin_prefetch(kbarF + (((size_t)(bb * RR) * HH + h) * 64 +
                                    (size_t)(js + 1) * 2) * 512, 0, 1);
    }
#if USE_TDM
    if ((tid >> 5) == 0) {
      if (js + 1 < NN / 32) __builtin_amdgcn_s_wait_tensorcnt(2);
      else                  __builtin_amdgcn_s_wait_tensorcnt(0);
    }
#endif
    __syncthreads();
    const float* aT = adjT[cur];
    const float* eT = erelT[cur];
#pragma unroll
    for (int jt = 0; jt < 2; ++jt) {
      v8f acc = {};
#pragma unroll
      for (int r = 0; r < RR; ++r) {
        v16bf kb = ldfrag(kbarF + (((size_t)(bb * RR + r) * HH + h) * 64 +
                                   (size_t)js * 2 + jt) * 512);
        v8f z = {};
        v8f d = wmma_bf16(qf, kb, z);
#pragma unroll
        for (int v = 0; v < 8; ++v) {
          int idx = (v + hi * 8) * 32 + n + 16 * jt;
          acc[v] += eT[idx * RR + r] * pri[r] * d[v];
        }
      }
#pragma unroll
      for (int v = 0; v < 8; ++v) {
        int idx = (v + hi * 8) * 32 + n + 16 * jt;
        float xv = aT[idx] * acc[v];
        rs[v] += red16_add(xv);
        float nm = fmaxf(mx[v], red16_max(xv));
        sm[v] = sm[v] * __expf(mx[v] - nm) + red16_add(__expf(xv - nm));
        mx[v] = nm;
      }
    }
    __syncthreads();
  }
  if ((l & 15) == 0) {
#pragma unroll
    for (int v = 0; v < 8; ++v) hsum[h][v + hi * 8] = rs[v];
  }
  __syncthreads();
  if (tid < 16) {
    float s = 0.f;
    for (int hh = 0; hh < HH; ++hh) s += hsum[hh][tid];
    rok[tid] = (s > 1e-6f) ? 1.f : 0.f;
  }
  __syncthreads();
  float rcp[8];
#pragma unroll
  for (int v = 0; v < 8; ++v)
    rcp[v] = (sm[v] > 0.f) ? (rok[v + hi * 8] / sm[v]) : 0.f;

  // ---- pass 2: attn * score * Vsr aggregation ----
  v8f agg0 = {}, agg1 = {};
#if USE_TDM
  if ((tid >> 5) == 0) ISSUE_TILES(0, 0);
#else
  STAGE_FB(0, 0);
#endif
  for (int js = 0; js < NN / 32; ++js) {
    int cur = js & 1;
    if (js + 1 < NN / 32) {
#if USE_TDM
      if ((tid >> 5) == 0) ISSUE_TILES((js + 1) & 1, js + 1);
#else
      STAGE_FB((js + 1) & 1, js + 1);
#endif
      if (l == 0)
        __builtin_prefetch(ksrF + (((size_t)((bb * TT) * RR) * HH + h) * 64 +
                                   (size_t)(js + 1) * 2) * 512, 0, 1);
    }
    if (tid < 32 * TT)
      ts[tid / TT][tid % TT] =
          types[((size_t)bb * NN + (size_t)js * 32 + tid / TT) * TT + tid % TT];
#if USE_TDM
    if ((tid >> 5) == 0) {
      if (js + 1 < NN / 32) __builtin_amdgcn_s_wait_tensorcnt(2);
      else                  __builtin_amdgcn_s_wait_tensorcnt(0);
    }
#endif
    __syncthreads();
    const float* aT = adjT[cur];
    const float* eT = erelT[cur];
    float P[2][8];
#pragma unroll
    for (int jt = 0; jt < 2; ++jt) {
      v8f acc = {};
#pragma unroll
      for (int r = 0; r < RR; ++r) {
        v16bf kb = ldfrag(kbarF + (((size_t)(bb * RR + r) * HH + h) * 64 +
                                   (size_t)js * 2 + jt) * 512);
        v8f z = {};
        v8f d = wmma_bf16(qf, kb, z);
#pragma unroll
        for (int v = 0; v < 8; ++v) {
          int idx = (v + hi * 8) * 32 + n + 16 * jt;
          acc[v] += eT[idx * RR + r] * pri[r] * d[v];
        }
      }
#pragma unroll
      for (int v = 0; v < 8; ++v) {
        int idx = (v + hi * 8) * 32 + n + 16 * jt;
        P[jt][v] = __expf(aT[idx] * acc[v] - mx[v]) * rcp[v];
      }
    }
#pragma unroll 1
    for (int s = 0; s < TT; ++s) {
#pragma unroll 1
      for (int r = 0; r < RR; ++r) {
#pragma unroll
        for (int jt = 0; jt < 2; ++jt) {
          v16bf kf = ldfrag(ksrF + (((size_t)((bb * TT + s) * RR + r) * HH + h) * 64 +
                                    (size_t)js * 2 + jt) * 512);
          v8f z = {};
          v8f d = wmma_bf16(qf, kf, z);
#pragma unroll
          for (int v = 0; v < 8; ++v) {
            int m = v + hi * 8, jl = n + 16 * jt, idx = m * 32 + jl;
            float w = aT[idx] * eT[idx * RR + r] * pri[r] * ts[jl][s] * d[v];
            pw[h][idx] = P[jt][v] * w;
          }
        }
        lds_fence();
        v16bf pa = gatherA_f32(pw[h], 32);
        const __bf16* vb = vsrF + ((((size_t)((bb * TT + s) * RR + r) * HH + h) * 32 +
                                    js) * 2) * 512;
        agg0 = wmma_bf16(pa, ldfrag(vb), agg0);
        agg1 = wmma_bf16(pa, ldfrag(vb + 512), agg1);
        lds_fence();
      }
    }
    __syncthreads();
  }
#pragma unroll
  for (int v = 0; v < 8; ++v) {
    size_t row = (size_t)i0 + v + hi * 8;
    aggG[((size_t)bb * NN + row) * DOUT + h * DKK + n] = agg0[v];
    aggG[((size_t)bb * NN + row) * DOUT + h * DKK + 16 + n] = agg1[v];
  }
#undef ISSUE_TILES
#undef STAGE_FB
}

// ---------------------------------------------------------------------------
// Kernel 4: gelu -> per-type Wa GEMM -> gated residual -> LayerNorm -> type mix
// grid(64,B), 256 thr (8 waves, each wave 2 o-tiles)
// ---------------------------------------------------------------------------
__global__ __launch_bounds__(256)
void k_out(const float* __restrict__ aggG, const float* __restrict__ x,
           const float* __restrict__ types,
           const float* __restrict__ Wa, const float* __restrict__ ba,
           const float* __restrict__ skip,
           const float* __restrict__ gamma, const float* __restrict__ beta,
           float* __restrict__ out) {
  __shared__ __bf16 gl[16 * DOUT];
  __shared__ float res[TT][16][DOUT];
  __shared__ float mus[TT][16], rsd[TT][16];
  int bb = blockIdx.y, n0 = blockIdx.x * 16;
  int tid = threadIdx.x;
  for (int i = tid; i < 16 * DOUT; i += 256) {
    float a = aggG[((size_t)bb * NN + n0 + (i >> 8)) * DOUT + (i & 255)];
    gl[i] = (__bf16)(0.5f * a * (1.f + erff(a * 0.70710678118654752f)));
  }
  __syncthreads();
  int wv = tid >> 5;
  int l = lane_id(), n = l & 15, hi = l >> 4;
#pragma unroll 1
  for (int oi = 0; oi < 2; ++oi) {
    int o0 = (wv * 2 + oi) * 16;
#pragma unroll 1
    for (int t = 0; t < TT; ++t) {
      v8f c = {};
#pragma unroll
      for (int kc = 0; kc < 8; ++kc) {
        v16bf a = gatherA_bf(gl + kc * 32, DOUT);
        v16bf b = gatherB_f32(Wa + (size_t)t * DOUT * DOUT + (size_t)kc * 32 * DOUT + o0, DOUT);
        c = wmma_bf16(a, b, c);
      }
      float alpha = 1.f / (1.f + __expf(-skip[t]));
      float bias = ba[t * DOUT + o0 + n];
#pragma unroll
      for (int v = 0; v < 8; ++v) {
        int m = v + hi * 8;
        float xv = x[((size_t)bb * NN + n0 + m) * DIN + o0 + n];
        res[t][m][o0 + n] = alpha * (c[v] + bias) + (1.f - alpha) * xv;
      }
    }
  }
  __syncthreads();
  if (tid < TT * 16) {
    int t = tid >> 4, row = tid & 15;
    float s = 0.f, s2 = 0.f;
    for (int o = 0; o < DOUT; ++o) { float v = res[t][row][o]; s += v; s2 += v * v; }
    float mu = s / DOUT;
    mus[t][row] = mu;
    rsd[t][row] = rsqrtf(s2 / DOUT - mu * mu + 1e-5f);
  }
  __syncthreads();
  for (int i = tid; i < 16 * DOUT; i += 256) {
    int row = i >> 8, col = i & 255;
    float o = 0.f;
#pragma unroll
    for (int t = 0; t < TT; ++t) {
      float v = (res[t][row][col] - mus[t][row]) * rsd[t][row] * gamma[t * DOUT + col] +
                beta[t * DOUT + col];
      o += types[((size_t)bb * NN + n0 + row) * TT + t] * v;
    }
    out[((size_t)bb * NN + n0 + row) * DOUT + col] = o;
  }
}

// ---------------------------------------------------------------------------
extern "C" void kernel_launch(void* const* d_in, const int* in_sizes, int n_in,
                              void* d_out, int out_size, void* d_ws, size_t ws_size,
                              hipStream_t stream) {
  (void)in_sizes; (void)n_in; (void)out_size; (void)ws_size;
  const float* x     = (const float*)d_in[0];
  const float* types = (const float*)d_in[1];
  const float* adj   = (const float*)d_in[2];
  const float* erel  = (const float*)d_in[3];
  const float* Wq    = (const float*)d_in[4];
  const float* bq    = (const float*)d_in[5];
  const float* Wk    = (const float*)d_in[6];
  const float* bk    = (const float*)d_in[7];
  const float* Wv    = (const float*)d_in[8];
  const float* bv    = (const float*)d_in[9];
  const float* Wa    = (const float*)d_in[10];
  const float* ba    = (const float*)d_in[11];
  const float* pri   = (const float*)d_in[12];
  const float* ratt  = (const float*)d_in[13];
  const float* rmsg  = (const float*)d_in[14];
  const float* skip  = (const float*)d_in[15];
  const float* gam   = (const float*)d_in[16];
  const float* bet   = (const float*)d_in[17];
  float* out = (float*)d_out;

  char* p = (char*)d_ws;
  const size_t szMat = (size_t)NB * TT * NN * DOUT;  // f32 elems
  float*  qt    = (float*)p;  p += szMat * 4;
  float*  kt    = (float*)p;  p += szMat * 4;
  float*  vt    = (float*)p;  p += szMat * 4;
  __bf16* qbarF = (__bf16*)p; p += (size_t)NB * HH * 64 * 512 * 2;
  __bf16* kbarF = (__bf16*)p; p += (size_t)NB * RR * HH * 64 * 512 * 2;
  __bf16* ksrF  = (__bf16*)p; p += (size_t)NB * TT * RR * HH * 64 * 512 * 2;
  __bf16* vsrF  = (__bf16*)p; p += (size_t)NB * TT * RR * HH * 32 * 2 * 512 * 2;
  float*  aggG  = (float*)p;  p += (size_t)NB * NN * DOUT * 4;

  k_qkv<<<dim3(NN / 16, NB), 512, 0, stream>>>(x, Wq, bq, Wk, bk, Wv, bv, qt, kt, vt);
  k_relmix<<<dim3(NN / 32, NB), 256, 0, stream>>>(qt, kt, vt, types, ratt, rmsg,
                                                  qbarF, kbarF, ksrF, vsrF);
  k_attn<<<dim3(NN / 16, NB), 256, 0, stream>>>(adj, erel, types, pri,
                                                qbarF, kbarF, ksrF, vsrF, aggG);
  k_out<<<dim3(NN / 16, NB), 256, 0, stream>>>(aggG, x, types, Wa, ba, skip, gam, bet, out);
}
